// HierarchicalCADNet_46316927320587
// MI455X (gfx1250) — matrix operand order repaired and checked by reading, not compiled, not run on hardware
//
#include <hip/hip_runtime.h>
#include <hip/hip_bf16.h>
#include <stdint.h>

typedef __bf16 bf16_t;
typedef __attribute__((ext_vector_type(16))) __bf16 v16bf;
typedef __attribute__((ext_vector_type(8)))  float  v8f;

#define DN1 8192
#define DN2 2048
#define FDIM 8
#define UDIM 128
#define CDIM 25
#define LNUM 7
#define TK    32   // K-step per staged tile
#define PITCH 40   // LDS row pitch in bf16 elements (32 data + 8 pad = 80B, conflict-free)

// ---------------------------------------------------------------- conversions
__global__ __launch_bounds__(256) void k_cvt_bf16(const float* __restrict__ s,
                                                  bf16_t* __restrict__ d, long n) {
  long i = (long)blockIdx.x * blockDim.x + threadIdx.x;
  long st = (long)gridDim.x * blockDim.x;
  for (; i < n; i += st) d[i] = (bf16_t)s[i];
}

// d[c*R + r] = (bf16) s[r*C + c]   (R, C multiples of 32)
__global__ __launch_bounds__(256) void k_transpose_cvt(const float* __restrict__ s,
                                                       bf16_t* __restrict__ d,
                                                       int R, int Ccols) {
  __shared__ bf16_t tile[32][33];
  int c0 = blockIdx.x * 32, r0 = blockIdx.y * 32;
  int tx = threadIdx.x & 31, ty = threadIdx.x >> 5;
  #pragma unroll
  for (int i = ty; i < 32; i += 8)
    tile[i][tx] = (bf16_t)s[(long)(r0 + i) * Ccols + (c0 + tx)];
  __syncthreads();
  #pragma unroll
  for (int i = ty; i < 32; i += 8)
    d[(long)(c0 + i) * R + (r0 + tx)] = tile[tx][i];
}

// ---------------------------------------------------------------- WMMA GEMM
// C[M,128] = A[M,K](bf16,row-major,lda) @ B[K,128], with B given transposed as
// BT[128,K] (row pitch K). Output fp32 (Cf, ld 128) and/or bf16 (Cb, ld ldcb).
// Async global->LDS staging (ASYNCcnt) with double-buffered LDS tiles.
__device__ __forceinline__ v16bf load_frag_a(const bf16_t* p) {
  union { uint4 u[2]; v16bf v; } t;
  t.u[0] = *(const uint4*)(p);
  t.u[1] = *(const uint4*)(p + 16);
  return t.v;
}
__device__ __forceinline__ v16bf load_frag_b(const bf16_t* p) {
  union { uint4 u[2]; v16bf v; } t;
  t.u[0] = *(const uint4*)(p);
  t.u[1] = *(const uint4*)(p + 16);
  return t.v;
}

// one b128 async copy: LDS[lds_off] <- *(sbase + gvoff)
__device__ __forceinline__ void async_b128(unsigned lds_off, int gvoff, const void* sbase) {
  asm volatile("global_load_async_to_lds_b128 %0, %1, %2"
               :: "v"(lds_off), "v"(gvoff), "s"(sbase) : "memory");
}
#define WAIT_ASYNC(n) asm volatile("s_wait_asynccnt " #n ::: "memory")

__global__ __launch_bounds__(256) void k_gemm_wmma(const bf16_t* __restrict__ A, int lda,
                                                   const bf16_t* __restrict__ BT, int K,
                                                   float* __restrict__ Cf,
                                                   bf16_t* __restrict__ Cb, int ldcb) {
  __shared__ bf16_t sA[2 * 128 * PITCH];
  __shared__ bf16_t sB[2 * 128 * PITCH];

  const int tid  = threadIdx.x;
  const int lane = tid & 31;
  const int wave = tid >> 5;          // 8 waves
  const int wm   = wave & 3;          // 4 M-groups of 32 rows
  const int wn   = wave >> 2;         // 2 N-groups of 64 cols
  const long mblock = (long)blockIdx.x * 128;

  v8f acc[2][4] = {};

  const int frow = lane & 15;         // row within 16x16 tile
  const int aoff = (lane >> 4) * 8;   // A frag: K+0..7/16..23 vs K+8..15/24..31
  const int boff = (lane >> 4) * 8;   // B frag: K+0..7/16..23 vs K+8..15/24..31
  const int srow = tid >> 2;          // staging row 0..63 (x2 iters)
  const int sce  = (tid & 3) * 8;     // staging col (elements, 16B chunks)

  // per-thread constant parts of global byte offsets (fit in i32)
  const int gaBase0 = (int)(((mblock + srow) * (long)lda + sce) * 2);
  const int gaBase1 = (int)(((mblock + srow + 64) * (long)lda + sce) * 2);
  const int gbBase0 = (srow * K + sce) * 2;
  const int gbBase1 = ((srow + 64) * K + sce) * 2;
  const unsigned laBase0 = (unsigned)(uintptr_t)(sA + srow * PITCH + sce);
  const unsigned laBase1 = (unsigned)(uintptr_t)(sA + (srow + 64) * PITCH + sce);
  const unsigned lbBase0 = (unsigned)(uintptr_t)(sB + srow * PITCH + sce);
  const unsigned lbBase1 = (unsigned)(uintptr_t)(sB + (srow + 64) * PITCH + sce);
  const unsigned bufBytes = 128 * PITCH * 2;

  const int ntiles = K / TK;

  // prologue: tile 0 -> buffer 0 (4 async instructions per wave per tile)
  {
    int k2 = 0;
    async_b128(laBase0, gaBase0 + k2, A);
    async_b128(lbBase0, gbBase0 + k2, BT);
    async_b128(laBase1, gaBase1 + k2, A);
    async_b128(lbBase1, gbBase1 + k2, BT);
  }

  for (int t = 0; t < ntiles; ++t) {
    const int cur = t & 1;
    if (t + 1 < ntiles) {
      const unsigned bo = (cur ^ 1) ? bufBytes : 0u;
      const int k2 = (t + 1) * TK * 2;
      async_b128(laBase0 + bo, gaBase0 + k2, A);
      async_b128(lbBase0 + bo, gbBase0 + k2, BT);
      async_b128(laBase1 + bo, gaBase1 + k2, A);
      async_b128(lbBase1 + bo, gbBase1 + k2, BT);
      WAIT_ASYNC(4);   // in-order returns: tile t fully resident
    } else {
      WAIT_ASYNC(0);
    }
    __syncthreads();

    const bf16_t* pA = sA + cur * (128 * PITCH);
    const bf16_t* pB = sB + cur * (128 * PITCH);
    v16bf af[2], bfv[4];
    #pragma unroll
    for (int i = 0; i < 2; i++)
      af[i] = load_frag_a(pA + (wm * 32 + i * 16 + frow) * PITCH + aoff);
    #pragma unroll
    for (int j = 0; j < 4; j++)
      bfv[j] = load_frag_b(pB + (wn * 64 + j * 16 + frow) * PITCH + boff);
    #pragma unroll
    for (int i = 0; i < 2; i++)
      #pragma unroll
      for (int j = 0; j < 4; j++)
        acc[i][j] = __builtin_amdgcn_wmma_f32_16x16x32_bf16(
            false, af[i], false, bfv[j], (short)0, acc[i][j], false, false);
    __syncthreads();
  }

  const int erow = (lane >> 4) * 8;
  const int ecol = lane & 15;
  #pragma unroll
  for (int i = 0; i < 2; i++)
    #pragma unroll
    for (int j = 0; j < 4; j++) {
      long rbase = mblock + wm * 32 + i * 16 + erow;
      int  col   = wn * 64 + j * 16 + ecol;
      #pragma unroll
      for (int e = 0; e < 8; e++) {
        float v = acc[i][j][e];
        if (Cf) Cf[(rbase + e) * UDIM + col] = v;
        if (Cb) Cb[(rbase + e) * (long)ldcb + col] = (bf16_t)v;
      }
    }
}

// ------------------------------------------------------- small dense helpers
__global__ __launch_bounds__(256) void k_gemm_k8(const float* __restrict__ V,
                                                 const float* __restrict__ W,
                                                 float* __restrict__ r) {
  int idx = blockIdx.x * 256 + threadIdx.x;
  int m = idx >> 7, n = idx & 127;
  float s = 0.f;
  #pragma unroll
  for (int k = 0; k < FDIM; k++) s += V[m * FDIM + k] * W[k * UDIM + n];
  r[idx] = s;
}

__global__ __launch_bounds__(256) void k_xw4(const float* __restrict__ x,
                                             const float* __restrict__ W,
                                             float* __restrict__ t) {
  int idx = blockIdx.x * 256 + threadIdx.x;
  int m = idx >> 2, j = idx & 3;
  float s = 0.f;
  for (int k = 0; k < UDIM; k++) s += x[m * UDIM + k] * W[k * 4 + j];
  t[idx] = s;
}

__global__ __launch_bounds__(256) void k_matvec4(const float* __restrict__ A, int K,
                                                 const float* __restrict__ B,
                                                 float* __restrict__ out) {
  __shared__ float red[4][256];
  int row = blockIdx.x, tid = threadIdx.x;
  float a[4] = {0.f, 0.f, 0.f, 0.f};
  for (int k = tid; k < K; k += 256) {
    float av = A[(long)row * K + k];
    #pragma unroll
    for (int j = 0; j < 4; j++) a[j] += av * B[k * 4 + j];
  }
  #pragma unroll
  for (int j = 0; j < 4; j++) red[j][tid] = a[j];
  __syncthreads();
  for (int s = 128; s > 0; s >>= 1) {
    if (tid < s)
      #pragma unroll
      for (int j = 0; j < 4; j++) red[j][tid] += red[j][tid + s];
    __syncthreads();
  }
  if (tid < 4) out[row * 4 + tid] = red[tid][0];
}

__global__ __launch_bounds__(256) void k_small_k4(const float* __restrict__ yf,
                                                  const float* __restrict__ x4,
                                                  const float* __restrict__ W1,
                                                  const float* __restrict__ W0,
                                                  float* __restrict__ r) {
  int idx = blockIdx.x * 256 + threadIdx.x;
  int m = idx >> 7, n = idx & 127;
  float s = 0.f;
  #pragma unroll
  for (int k = 0; k < 4; k++)
    s += yf[m * 4 + k] * W1[k * UDIM + n] + x4[m * 4 + k] * W0[k * UDIM + n];
  r[idx] = s;
}

__global__ __launch_bounds__(256) void k_pack_wt(const float* Wa, const float* Wb,
                                                 const float* Wc, const float* Wd,
                                                 int nblk, bf16_t* __restrict__ out) {
  int total = UDIM * nblk * UDIM;
  int idx = blockIdx.x * 256 + threadIdx.x;
  if (idx >= total) return;
  int Kc = nblk * UDIM;
  int n = idx / Kc;
  int k = idx - n * Kc;
  int b = k >> 7, kk = k & 127;
  const float* W = (b == 0) ? Wa : (b == 1) ? Wb : (b == 2) ? Wc : Wd;
  out[idx] = (bf16_t)W[kk * UDIM + n];
}

// -------------------------------------------------------------- batch norm
__global__ void k_zero256(float* p) { p[threadIdx.x] = 0.f; }

__global__ __launch_bounds__(256) void k_colstats(const float* __restrict__ r, int M,
                                                  int Ccols, float* __restrict__ stats) {
  int c = threadIdx.x % Ccols;
  int r0 = threadIdx.x / Ccols;
  int stride = 256 / Ccols;
  int rowBase = blockIdx.x * stride * 16;
  float s = 0.f, s2 = 0.f;
  #pragma unroll
  for (int i = 0; i < 16; i++) {
    int row = rowBase + i * stride + r0;
    if (row < M) {
      float v = r[(long)row * Ccols + c];
      s += v; s2 += v * v;
    }
  }
  atomicAdd(&stats[c], s);
  atomicAdd(&stats[Ccols + c], s2);
}

__global__ __launch_bounds__(256) void k_bn_apply(const float* __restrict__ r,
                                                  const float* __restrict__ stats,
                                                  const float* __restrict__ g,
                                                  const float* __restrict__ b,
                                                  const float* __restrict__ prev,
                                                  float* __restrict__ xout,
                                                  bf16_t* __restrict__ xb, int ldxb,
                                                  long n, int Ccols, float invM) {
  long idx = (long)blockIdx.x * 256 + threadIdx.x;
  if (idx >= n) return;
  int c = (int)(idx % Ccols);
  long row = idx / Ccols;
  float mean = stats[c] * invM;
  float var  = stats[Ccols + c] * invM - mean * mean;
  float y = g[c] * (r[idx] - mean) * rsqrtf(var + 1e-5f) + b[c];
  y = fmaxf(y, 0.f);
  if (prev) y += prev[idx];
  xout[idx] = y;
  if (xb) xb[row * (long)ldxb + c] = (bf16_t)y;
}

// ---------------------------------------------------------------- softmax
__global__ __launch_bounds__(256) void k_softmax(const float* __restrict__ x,
                                                 const float* __restrict__ W,
                                                 const float* __restrict__ b,
                                                 float* __restrict__ out) {
  int row = blockIdx.x * 256 + threadIdx.x;
  if (row >= DN1) return;
  float lg[CDIM];
  #pragma unroll
  for (int j = 0; j < CDIM; j++) lg[j] = b[j];
  for (int k = 0; k < UDIM; k++) {
    float xv = x[(long)row * UDIM + k];
    #pragma unroll
    for (int j = 0; j < CDIM; j++) lg[j] += xv * W[k * CDIM + j];
  }
  float mx = lg[0];
  #pragma unroll
  for (int j = 1; j < CDIM; j++) mx = fmaxf(mx, lg[j]);
  float s = 0.f;
  #pragma unroll
  for (int j = 0; j < CDIM; j++) { lg[j] = __expf(lg[j] - mx); s += lg[j]; }
  float inv = 1.f / s;
  #pragma unroll
  for (int j = 0; j < CDIM; j++) out[(long)row * CDIM + j] = lg[j] * inv;
}

// ================================================================== launch
extern "C" void kernel_launch(void* const* d_in, const int* in_sizes, int n_in,
                              void* d_out, int out_size, void* d_ws, size_t ws_size,
                              hipStream_t stream) {
  (void)in_sizes; (void)n_in; (void)out_size; (void)ws_size;
  const float* V1 = (const float*)d_in[0];
  const float* E1 = (const float*)d_in[1];
  const float* E2 = (const float*)d_in[2];
  const float* E3 = (const float*)d_in[3];
  const float* V2 = (const float*)d_in[4];
  const float* A2 = (const float*)d_in[5];
  const float* A3 = (const float*)d_in[6];
  const float* ge_start_W = (const float*)d_in[7];
  const float* bn_start_g = (const float*)d_in[9];
  const float* bn_start_b = (const float*)d_in[10];
  const float* a4_W    = (const float*)d_in[11];
  const float* bn_a4_g = (const float*)d_in[13];
  const float* bn_a4_b = (const float*)d_in[14];
  const float* g2_W1f  = (const float*)d_in[15];
  const float* g2_W0f  = (const float*)d_in[16];
  const float* g2_W1   = (const float*)d_in[18];
  const float* g2_W0   = (const float*)d_in[19];
  const float* bn2_g   = (const float*)d_in[21];
  const float* bn2_b   = (const float*)d_in[22];
  const float* ge2_W   = (const float*)d_in[23];
  const float* bn_2_g  = (const float*)d_in[25];
  const float* bn_2_b  = (const float*)d_in[26];
  const float* a3_W    = (const float*)d_in[27];
  const float* bn_a3_g = (const float*)d_in[29];
  const float* bn_a3_b = (const float*)d_in[30];
  const float* g1_W1   = (const float*)d_in[31];
  const float* g1_W2   = (const float*)d_in[32];
  const float* g1_W3   = (const float*)d_in[33];
  const float* g1_W0   = (const float*)d_in[34];
  const float* bn1_g   = (const float*)d_in[36];
  const float* bn1_b   = (const float*)d_in[37];
  const float* ge1_W   = (const float*)d_in[38];
  const float* bn_1_g  = (const float*)d_in[40];
  const float* bn_1_b  = (const float*)d_in[41];
  const float* gef_W   = (const float*)d_in[42];
  const float* gef_b   = (const float*)d_in[43];
  float* out = (float*)d_out;

  char* wp = (char*)d_ws;
  auto alloc = [&](size_t bytes) -> void* {
    uintptr_t q = ((uintptr_t)wp + 255) & ~(uintptr_t)255;
    wp = (char*)(q + bytes);
    return (void*)q;
  };
  bf16_t* E1b   = (bf16_t*)alloc((size_t)DN1 * DN1 * 2);
  bf16_t* E2b   = (bf16_t*)alloc((size_t)DN1 * DN1 * 2);
  bf16_t* E3b   = (bf16_t*)alloc((size_t)DN1 * DN1 * 2);
  bf16_t* A2b   = (bf16_t*)alloc((size_t)DN2 * DN2 * 2);
  bf16_t* A3Tb  = (bf16_t*)alloc((size_t)DN1 * DN2 * 2);  // [8192,2048]
  bf16_t* Ycat1 = (bf16_t*)alloc((size_t)DN1 * 512 * 2);
  bf16_t* Ycat2 = (bf16_t*)alloc((size_t)DN2 * 256 * 2);
  float*  x1    = (float*)alloc((size_t)DN1 * UDIM * 4);
  bf16_t* x1T   = (bf16_t*)alloc((size_t)UDIM * DN1 * 2);
  float*  x2    = (float*)alloc((size_t)DN2 * UDIM * 4);
  bf16_t* x2T   = (bf16_t*)alloc((size_t)UDIM * DN2 * 2);
  float*  r1    = (float*)alloc((size_t)DN1 * UDIM * 4);
  float*  r2    = (float*)alloc((size_t)DN2 * UDIM * 4);
  float*  t4    = (float*)alloc((size_t)DN1 * 4 * 4);
  float*  a4r   = (float*)alloc((size_t)DN2 * 4 * 4);
  float*  x24   = (float*)alloc((size_t)DN2 * 4 * 4);
  float*  yf    = (float*)alloc((size_t)DN2 * 4 * 4);
  float*  t2    = (float*)alloc((size_t)DN2 * UDIM * 4);
  bf16_t* t2T   = (bf16_t*)alloc((size_t)UDIM * DN2 * 2);
  float*  x1f   = (float*)alloc((size_t)DN1 * UDIM * 4);
  bf16_t* Wct   = (bf16_t*)alloc((size_t)UDIM * 512 * 2);
  float*  stats = (float*)alloc(256 * 4);

  auto run_bn = [&](const float* r, int M, int Cc, const float* g, const float* b,
                    const float* prev, float* xout, bf16_t* xb, int ldxb) {
    k_zero256<<<1, 256, 0, stream>>>(stats);
    int stride = 256 / Cc, rpb = stride * 16;
    k_colstats<<<(M + rpb - 1) / rpb, 256, 0, stream>>>(r, M, Cc, stats);
    long n = (long)M * Cc;
    k_bn_apply<<<(int)((n + 255) / 256), 256, 0, stream>>>(r, stats, g, b, prev,
                                                           xout, xb, ldxb, n, Cc,
                                                           1.0f / (float)M);
  };

  // ---- one-time conversions (biases before BN cancel exactly; omitted)
  k_cvt_bf16<<<8192, 256, 0, stream>>>(E1, E1b, (long)DN1 * DN1);
  k_cvt_bf16<<<8192, 256, 0, stream>>>(E2, E2b, (long)DN1 * DN1);
  k_cvt_bf16<<<8192, 256, 0, stream>>>(E3, E3b, (long)DN1 * DN1);
  k_cvt_bf16<<<2048, 256, 0, stream>>>(A2, A2b, (long)DN2 * DN2);
  k_transpose_cvt<<<dim3(DN1 / 32, DN2 / 32), 256, 0, stream>>>(A3, A3Tb, DN2, DN1);

  // ---- x1 = relu(BN(V1 @ ge_start_W))
  k_gemm_k8<<<DN1 * UDIM / 256, 256, 0, stream>>>(V1, ge_start_W, r1);
  run_bn(r1, DN1, UDIM, bn_start_g, bn_start_b, nullptr, x1, Ycat1 + 384, 512);
  k_transpose_cvt<<<dim3(UDIM / 32, DN1 / 32), 256, 0, stream>>>(x1, x1T, DN1, UDIM);

  // ---- x2 = V2 + relu(BN(A3 @ (x1 @ a4_W)))
  k_xw4<<<DN1 * 4 / 256, 256, 0, stream>>>(x1, a4_W, t4);
  k_matvec4<<<DN2, 256, 0, stream>>>(A3, DN1, t4, a4r);
  run_bn(a4r, DN2, 4, bn_a4_g, bn_a4_b, V2, x24, nullptr, 0);

  // ---- level-2 first layer: x2 = relu(BN((A2@x2)@W1f + x2@W0f))
  k_matvec4<<<DN2, 256, 0, stream>>>(A2, DN2, x24, yf);
  k_small_k4<<<DN2 * UDIM / 256, 256, 0, stream>>>(yf, x24, g2_W1f, g2_W0f, r2);
  run_bn(r2, DN2, UDIM, bn2_g, bn2_b, nullptr, x2, Ycat2 + 128, 256);
  k_transpose_cvt<<<dim3(UDIM / 32, DN2 / 32), 256, 0, stream>>>(x2, x2T, DN2, UDIM);

  // ---- level-2 stack (residual layers)
  for (int i = 0; i < LNUM - 1; i++) {
    k_gemm_wmma<<<DN2 / 128, 256, 0, stream>>>(A2b, DN2, x2T, DN2,
                                               nullptr, Ycat2, 256);
    k_pack_wt<<<(UDIM * 256 + 255) / 256, 256, 0, stream>>>(
        g2_W1 + (size_t)i * UDIM * UDIM, g2_W0 + (size_t)i * UDIM * UDIM,
        nullptr, nullptr, 2, Wct);
    k_gemm_wmma<<<DN2 / 128, 256, 0, stream>>>(Ycat2, 256, Wct, 256,
                                               r2, nullptr, 0);
    run_bn(r2, DN2, UDIM, bn2_g + (i + 1) * UDIM, bn2_b + (i + 1) * UDIM,
           x2, x2, Ycat2 + 128, 256);
    k_transpose_cvt<<<dim3(UDIM / 32, DN2 / 32), 256, 0, stream>>>(x2, x2T, DN2, UDIM);
  }

  // ---- x2 = relu(BN(x2 @ ge2_W))
  k_pack_wt<<<(UDIM * UDIM + 255) / 256, 256, 0, stream>>>(ge2_W, nullptr, nullptr,
                                                           nullptr, 1, Wct);
  k_gemm_wmma<<<DN2 / 128, 256, 0, stream>>>(Ycat2 + 128, 256, Wct, UDIM,
                                             r2, nullptr, 0);
  run_bn(r2, DN2, UDIM, bn_2_g, bn_2_b, nullptr, x2, Ycat2 + 128, 256);

  // ---- x1 = x1 + relu(BN(A3^T @ (x2 @ a3_W)))
  k_pack_wt<<<(UDIM * UDIM + 255) / 256, 256, 0, stream>>>(a3_W, nullptr, nullptr,
                                                           nullptr, 1, Wct);
  k_gemm_wmma<<<DN2 / 128, 256, 0, stream>>>(Ycat2 + 128, 256, Wct, UDIM,
                                             t2, nullptr, 0);
  k_transpose_cvt<<<dim3(UDIM / 32, DN2 / 32), 256, 0, stream>>>(t2, t2T, DN2, UDIM);
  k_gemm_wmma<<<DN1 / 128, 256, 0, stream>>>(A3Tb, DN2, t2T, DN2, r1, nullptr, 0);
  run_bn(r1, DN1, UDIM, bn_a3_g, bn_a3_b, x1, x1, Ycat1 + 384, 512);
  k_transpose_cvt<<<dim3(UDIM / 32, DN1 / 32), 256, 0, stream>>>(x1, x1T, DN1, UDIM);

  // ---- level-1 stack: x1 += relu(BN(E1 x W1 + E2 x W2 + E3 x W3 + x W0))
  for (int i = 0; i < LNUM; i++) {
    k_gemm_wmma<<<DN1 / 128, 256, 0, stream>>>(E1b, DN1, x1T, DN1,
                                               nullptr, Ycat1 + 0, 512);
    k_gemm_wmma<<<DN1 / 128, 256, 0, stream>>>(E2b, DN1, x1T, DN1,
                                               nullptr, Ycat1 + 128, 512);
    k_gemm_wmma<<<DN1 / 128, 256, 0, stream>>>(E3b, DN1, x1T, DN1,
                                               nullptr, Ycat1 + 256, 512);
    size_t wo = (size_t)i * UDIM * UDIM;
    k_pack_wt<<<(UDIM * 512 + 255) / 256, 256, 0, stream>>>(
        g1_W1 + wo, g1_W2 + wo, g1_W3 + wo, g1_W0 + wo, 4, Wct);
    k_gemm_wmma<<<DN1 / 128, 256, 0, stream>>>(Ycat1, 512, Wct, 512,
                                               r1, nullptr, 0);
    run_bn(r1, DN1, UDIM, bn1_g + i * UDIM, bn1_b + i * UDIM,
           x1, x1, Ycat1 + 384, 512);
    k_transpose_cvt<<<dim3(UDIM / 32, DN1 / 32), 256, 0, stream>>>(x1, x1T, DN1, UDIM);
  }

  // ---- x1 = relu(BN(x1 @ ge1_W)); out = softmax(x1 @ gef_W + gef_b)
  k_pack_wt<<<(UDIM * UDIM + 255) / 256, 256, 0, stream>>>(ge1_W, nullptr, nullptr,
                                                           nullptr, 1, Wct);
  k_gemm_wmma<<<DN1 / 128, 256, 0, stream>>>(Ycat1 + 384, 512, Wct, UDIM,
                                             r1, nullptr, 0);
  run_bn(r1, DN1, UDIM, bn_1_g, bn_1_b, nullptr, x1f, nullptr, 0);
  k_softmax<<<(DN1 + 255) / 256, 256, 0, stream>>>(x1f, gef_W, gef_b, out);
}
